// Simple_LSTM_36996848288364
// MI455X (gfx1250) — 16-bit and 8-bit matrix operand order repaired and checked by reading, compiled, not run on hardware
//
#include <hip/hip_runtime.h>
#include <cstdint>
#include <cstddef>

#define B_ 512
#define T_ 512
#define F_ 128
#define U_ 256
#define G_ 1024

typedef int          v8i  __attribute__((ext_vector_type(8)));
typedef float        v8f  __attribute__((ext_vector_type(8)));
typedef _Float16     v16h __attribute__((ext_vector_type(16)));
typedef unsigned int v4u  __attribute__((ext_vector_type(4)));
typedef int          v4i  __attribute__((ext_vector_type(4)));

static constexpr size_t XZ_OFF    = 0;
static constexpr size_t KPACK_OFF = (size_t)T_ * B_ * G_ * 2;
static constexpr size_t RPACK_OFF = KPACK_OFF + 256 * 1024;
static constexpr size_t H_OFF     = RPACK_OFF + 256 * 1024;

#define WSCALE 256.0f
#define HSCALE 16.0f
#define INV_ACC_SCALE (1.0f / (WSCALE * HSCALE))

__device__ __forceinline__ uint8_t f32_to_fp8_e4m3(float x) {
    union { float f; uint32_t u; } v;
    v.f = x;
    uint32_t s = (v.u >> 31) & 1u;
    v.u &= 0x7FFFFFFFu;
    float ax = v.f;
    uint8_t code;
    if (!(ax > 0.0f)) {
        code = 0;
    } else {
        int   e    = (int)((v.u >> 23) & 0xFF) - 127;
        uint32_t m = v.u & 0x7FFFFFu;
        if (e >= -6) {
            uint32_t mm  = m >> 20;
            uint32_t rem = m & 0xFFFFFu;
            if (rem > 0x80000u || (rem == 0x80000u && (mm & 1u))) mm++;
            int e8 = e + 7;
            if (mm == 8u) { mm = 0u; e8++; }
            if (e8 >= 16) { e8 = 15; mm = 6; }
            code = (uint8_t)((e8 << 3) | mm);
        } else {
            float scaled = ax * 512.0f;
            int mm = (int)(scaled + 0.5f);
            code = (mm > 7) ? (uint8_t)(1u << 3) : (uint8_t)mm;
        }
    }
    return (uint8_t)(code | (s << 7));
}

__device__ __forceinline__ void tdm_load_1d(const void* gptr, uint32_t lds_byte_off,
                                            uint32_t n8) {
    uint64_t ga = (uint64_t)(uintptr_t)gptr;
    v4u g0;
    g0[0] = 1u;
    g0[1] = lds_byte_off;
    g0[2] = (uint32_t)(ga & 0xFFFFFFFFu);
    g0[3] = (uint32_t)((ga >> 32) & 0x1FFFFFFu) | (2u << 30);
    v8i g1;
    g1[0] = (int)(3u << 16);
    g1[1] = (int)(n8 << 16);
    g1[2] = (int)((n8 >> 16) | (1u << 16));
    g1[3] = (int)(n8 << 16);
    g1[4] = 1;
    g1[5] = (int)n8;
    g1[6] = 0;
    g1[7] = 0;
    v4i gz4 = {0, 0, 0, 0};
    v8i gz8 = {0, 0, 0, 0, 0, 0, 0, 0};
    __builtin_amdgcn_tensor_load_to_lds(g0, g1, gz4, gz4, gz8, 0);
}

__global__ void pack_wk_f16(const float* __restrict__ wk, _Float16* __restrict__ dst) {
    int q      = blockIdx.x * 256 + threadIdx.x;
    int frag   = q >> 9;
    int within = q & 511;
    int lane   = within >> 4;
    int h      = within & 15;
    int v = h >> 1, j = h & 1;
    int kt = frag & 3, nt = frag >> 2;
    int K = (h < 8) ? ((lane >> 4) * 8 + h) : (16 + (lane >> 4) * 8 + (h - 8));
    int N = lane & 15;
    dst[(size_t)frag * 512 + lane * 16 + h] = (_Float16)wk[(size_t)(kt * 32 + K) * G_ + nt * 16 + N];
}

__global__ void pack_wr_fp8(const float* __restrict__ wr, uint8_t* __restrict__ dst) {
    int q      = blockIdx.x * 256 + threadIdx.x;
    int frag   = q >> 10;
    int within = q & 1023;
    int lane   = within >> 5;
    int b      = within & 31;
    int v = b >> 2, j = b & 3;
    int kt = frag & 3, nt = frag >> 2;
    int K = (v >> 1) * 16 + (lane >> 4) * 8 + (v & 1) * 4 + j;
    int N = lane & 15;
    dst[q] = f32_to_fp8_e4m3(wr[(size_t)(kt * 64 + K) * G_ + nt * 16 + N] * WSCALE);
}

__global__ __launch_bounds__(256) void proj_kernel(const float* __restrict__ x,
                                                   const float* __restrict__ bias,
                                                   const _Float16* __restrict__ kpack,
                                                   _Float16* __restrict__ xz) {
    __shared__ float xs[16 * 128];
    int tid = threadIdx.x, lane = tid & 31, w = tid >> 5;
    int mt = blockIdx.x;
    int t  = mt >> 5;
    int b0 = (mt & 31) << 4;

    for (int i = tid; i < 16 * 128; i += 256) {
        int row = i >> 7, col = i & 127;
        xs[i] = x[((size_t)(b0 + row) * T_ + t) * F_ + col];
    }
    __syncthreads();

    int grp = lane >> 4, li = lane & 15;
    v16h a[4];
    #pragma unroll
    for (int kt = 0; kt < 4; ++kt) {
        #pragma unroll
        for (int v = 0; v < 8; ++v)
            #pragma unroll
            for (int j = 0; j < 2; ++j) {
                int K = (v & 3) * 2 + (v >> 2) * 16 + grp * 8 + j;
                a[kt][v * 2 + j] = (_Float16)xs[li * 128 + kt * 32 + K];
            }
    }

    for (int ntl = 0; ntl < 8; ++ntl) {
        int nt = w * 8 + ntl;
        v8f acc = {};
        #pragma unroll
        for (int kt = 0; kt < 4; ++kt) {
            v16h bf = *(const v16h*)(kpack + (size_t)(nt * 4 + kt) * 512 + lane * 16);
            acc = __builtin_amdgcn_wmma_f32_16x16x32_f16(false, a[kt], false, bf,
                                                         (short)0, acc, false, false);
        }
        int col = nt * 16 + li;
        float bv = bias[col];
        #pragma unroll
        for (int r = 0; r < 8; ++r) {
            int M = grp * 8 + r;
            xz[((size_t)(t * B_ + b0 + M)) * G_ + col] = (_Float16)(acc[r] + bv);
        }
    }
}

#define LDS_ZOFF 262144u
#define LDS_HOFF (262144u + 32768u)
__global__ __launch_bounds__(256) void recur_kernel(const _Float16* __restrict__ xz,
                                                    const uint8_t* __restrict__ rpack,
                                                    float* __restrict__ hout) {
    extern __shared__ char smem[];
    uint8_t*  recw = (uint8_t*)smem;
    _Float16* zbuf = (_Float16*)(smem + LDS_ZOFF);
    uint8_t*  hbuf = (uint8_t*)(smem + LDS_HOFF);

    int tid = threadIdx.x, lane = tid & 31, w = tid >> 5;
    int b0 = blockIdx.x * 16;

    if (w == 0) tdm_load_1d(rpack, 0u, 262144u / 8u);
    for (int i = tid; i < 4096 / 4; i += 256) ((uint32_t*)hbuf)[i] = 0;
    float c[16];
    #pragma unroll
    for (int m = 0; m < 16; ++m) c[m] = 0.0f;
    if (w == 0) __builtin_amdgcn_s_wait_tensorcnt(0);
    __syncthreads();

    int grp = lane >> 4, li = lane & 15;
    int u    = tid;
    int ktu  = u >> 6;
    int K    = u & 63;
    int vv   = 2 * (K >> 4) + ((K >> 2) & 1);
    int grpA = (K >> 3) & 1;
    int jb   = K & 3;

    for (int t = 0; t < T_; ++t) {
        if (w == 0)
            tdm_load_1d(xz + (size_t)(t * B_ + b0) * G_, LDS_ZOFF, 32768u / 8u);
        if (t + 1 < T_)
            __builtin_prefetch(xz + (size_t)((t + 1) * B_ + b0) * G_ + tid * 64, 0, 1);

        v8i afrag[4];
        #pragma unroll
        for (int kt = 0; kt < 4; ++kt)
            afrag[kt] = *(const v8i*)(hbuf + kt * 1024 + lane * 32);

        if (w == 0) __builtin_amdgcn_s_wait_tensorcnt(0);
        __syncthreads();

        for (int ntl = 0; ntl < 8; ++ntl) {
            int nt = w * 8 + ntl;
            v8f acc = {};
            #pragma unroll
            for (int kt = 0; kt < 4; ++kt) {
                v8i bf = *(const v8i*)(recw + (nt * 4 + kt) * 1024 + lane * 32);
                acc = __builtin_amdgcn_wmma_f32_16x16x64_fp8_fp8(afrag[kt], bf,
                                                                 (short)0, acc, false, false);
            }
            int col = nt * 16 + li;
            #pragma unroll
            for (int r = 0; r < 8; ++r) {
                int M = grp * 8 + r;
                float z = acc[r] * INV_ACC_SCALE + (float)zbuf[M * 1024 + col];
                zbuf[M * 1024 + col] = (_Float16)z;
            }
        }
        __syncthreads();

        #pragma unroll
        for (int m = 0; m < 16; ++m) {
            float zi = (float)zbuf[m * 1024 + u];
            float zf = (float)zbuf[m * 1024 + 256 + u];
            float zg = (float)zbuf[m * 1024 + 512 + u];
            float zo = (float)zbuf[m * 1024 + 768 + u];
            float ig = 1.0f / (1.0f + __expf(-zi));
            float fg = 1.0f / (1.0f + __expf(-zf));
            float gg = zg > 0.0f ? zg : 0.0f;
            float og = 1.0f / (1.0f + __expf(-zo));
            float cn = fg * c[m] + ig * gg;
            c[m] = cn;
            float h = og * (cn > 0.0f ? cn : 0.0f);
            hbuf[ktu * 1024 + (grpA * 16 + m) * 32 + vv * 4 + jb] = f32_to_fp8_e4m3(h * HSCALE);
            if (t == T_ - 1) hout[(size_t)(b0 + m) * U_ + u] = h;
        }
        __syncthreads();
    }
}

__global__ void dense_kernel(const float* __restrict__ h, const float* __restrict__ dw,
                             const float* __restrict__ db, float* __restrict__ out) {
    int b = blockIdx.x * 256 + threadIdx.x;
    if (b >= B_) return;
    float s = db[0];
    #pragma unroll 4
    for (int u2 = 0; u2 < U_; ++u2) s += h[(size_t)b * U_ + u2] * dw[u2];
    out[b] = s;
}

extern "C" void kernel_launch(void* const* d_in, const int* in_sizes, int n_in,
                              void* d_out, int out_size, void* d_ws, size_t ws_size,
                              hipStream_t stream) {
    (void)in_sizes; (void)n_in; (void)out_size; (void)ws_size;
    const float* x    = (const float*)d_in[0];
    const float* wk   = (const float*)d_in[1];
    const float* wr   = (const float*)d_in[2];
    const float* bias = (const float*)d_in[3];
    const float* dw   = (const float*)d_in[4];
    const float* db   = (const float*)d_in[5];
    float* out = (float*)d_out;

    char* ws = (char*)d_ws;
    _Float16* xz    = (_Float16*)(ws + XZ_OFF);
    _Float16* kpack = (_Float16*)(ws + KPACK_OFF);
    uint8_t*  rpack = (uint8_t*)(ws + RPACK_OFF);
    float*    hbufg = (float*)(ws + H_OFF);

    pack_wk_f16<<<131072 / 256, 256, 0, stream>>>(wk, kpack);
    pack_wr_fp8<<<262144 / 256, 256, 0, stream>>>(wr, rpack);
    proj_kernel<<<(T_ * B_) / 16, 256, 0, stream>>>(x, bias, kpack, xz);

    const size_t lds_bytes = 262144 + 32768 + 4096;
    (void)hipFuncSetAttribute(reinterpret_cast<const void*>(recur_kernel),
                              hipFuncAttributeMaxDynamicSharedMemorySize, (int)lds_bytes);
    recur_kernel<<<B_ / 16, 256, lds_bytes, stream>>>(xz, rpack, hbufg);

    dense_kernel<<<(B_ + 255) / 256, 256, 0, stream>>>(hbufg, dw, db, out);
}
